// EquiformerQM9_88974542504661
// MI455X (gfx1250) — matrix operand order repaired and checked by reading, not compiled, not run on hardware
//
#include <hip/hip_runtime.h>
#include <hip/hip_bf16.h>
#include <math.h>

// Problem constants (match reference)
#define Bsz 64
#define Nn_ 256
#define Dd  128
#define Hh  4
#define DHd 32
#define KSEL 16
#define NTt 11
#define NOo 19
#define FFd 256
#define BN  (Bsz * Nn_)   // 16384

typedef _Float16 h16;
typedef _Float16 v16h __attribute__((ext_vector_type(16)));
typedef _Float16 v8h  __attribute__((ext_vector_type(8)));
typedef float    v8f  __attribute__((ext_vector_type(8)));
typedef unsigned int u32x4 __attribute__((ext_vector_type(4)));
typedef int          i32x4 __attribute__((ext_vector_type(4)));
typedef int          i32x8 __attribute__((ext_vector_type(8)));

#if defined(__has_builtin)
#if __has_builtin(__builtin_amdgcn_tensor_load_to_lds) && __has_builtin(__builtin_amdgcn_s_wait_tensorcnt)
#define HAVE_TDM 1
#endif
#endif

// ---------------------------------------------------------------------------
// 1) Token embedding: h = x @ W_emb + b_emb   (K=11, trivial; VALU)
// ---------------------------------------------------------------------------
__global__ void embed_kernel(const float* __restrict__ x,
                             const float* __restrict__ W_emb,
                             const float* __restrict__ b_emb,
                             float* __restrict__ h,
                             h16* __restrict__ hf16) {
    int row = blockIdx.x;          // 0..BN-1
    int d   = threadIdx.x;         // 0..127
    const float* xr = x + row * NTt;
    float acc = b_emb[d];
    #pragma unroll
    for (int t = 0; t < NTt; ++t)
        acc = fmaf(xr[t], W_emb[t * Dd + d], acc);
    h[row * Dd + d]    = acc;
    hf16[row * Dd + d] = (h16)acc;
}

// ---------------------------------------------------------------------------
// 2) Weight convert+transpose: W [K x N] f32 -> Wt [N x K] f16 (B^T, N-major)
// ---------------------------------------------------------------------------
__global__ void cvt_weight_kernel(const float* __restrict__ W,
                                  h16* __restrict__ Wt, int Kd, int Nd) {
    int idx = blockIdx.x * blockDim.x + threadIdx.x;
    if (idx >= Kd * Nd) return;
    int k = idx / Nd, n = idx % Nd;
    Wt[n * Kd + k] = (h16)W[k * Nd + n];
}

// ---------------------------------------------------------------------------
// 3) Adjacency bitmask build: adj1[b,i,j] = adj & pm & ~eye, packed 8 dwords/row
// ---------------------------------------------------------------------------
__global__ void adj1_bits_kernel(const unsigned char* __restrict__ adj,
                                 const unsigned char* __restrict__ mask,
                                 unsigned int* __restrict__ bits) {
    int idx = blockIdx.x * blockDim.x + threadIdx.x;   // (b*N+i)*8 + chunk
    if (idx >= BN * 8) return;
    int chunk = idx & 7;
    int row   = idx >> 3;          // b*N+i
    int b = row >> 8, i = row & 255;
    unsigned int w = 0;
    unsigned char mi = mask[b * Nn_ + i];
    const unsigned char* arow = adj + (size_t)row * Nn_;
    #pragma unroll
    for (int t = 0; t < 32; ++t) {
        int j = chunk * 32 + t;
        unsigned char a = arow[j] & mi & mask[b * Nn_ + j];
        if (a && j != i) w |= (1u << t);
    }
    bits[idx] = w;
}

// 2-hop: adj2row[i] = (OR_{j in adj1row[i]} adj1row[j]) & ~adj1row[i] & ~eye
__global__ void adj2_bits_kernel(const unsigned int* __restrict__ a1,
                                 unsigned int* __restrict__ a2) {
    int row = blockIdx.x * blockDim.x + threadIdx.x;   // b*N+i
    if (row >= BN) return;
    int b = row >> 8, i = row & 255;
    const unsigned int* base  = a1 + (size_t)b * Nn_ * 8;
    const unsigned int* myrow = base + (size_t)i * 8;
    unsigned int acc[8] = {0,0,0,0,0,0,0,0};
    unsigned int my[8];
    #pragma unroll
    for (int c = 0; c < 8; ++c) my[c] = myrow[c];
    for (int j = 0; j < Nn_; ++j) {
        if ((my[j >> 5] >> (j & 31)) & 1u) {
            const unsigned int* jr = base + (size_t)j * 8;
            #pragma unroll
            for (int c = 0; c < 8; ++c) acc[c] |= jr[c];
        }
    }
    #pragma unroll
    for (int c = 0; c < 8; ++c) {
        unsigned int eye = (c == (i >> 5)) ? (1u << (i & 31)) : 0u;
        a2[(size_t)row * 8 + c] = acc[c] & ~my[c] & ~eye;
    }
}

// ---------------------------------------------------------------------------
// 4) Top-K nearest valid neighbors (1/2-hop within radius, self included)
// ---------------------------------------------------------------------------
__global__ void topk_kernel(const float* __restrict__ pos,
                            const unsigned int* __restrict__ a1,
                            const unsigned int* __restrict__ a2,
                            int* __restrict__ idxK,
                            int* __restrict__ degK,
                            int* __restrict__ nmaskK) {
    int row = blockIdx.x * blockDim.x + threadIdx.x;   // b*N+i
    if (row >= BN) return;
    int b = row >> 8, i = row & 255;
    float px = pos[(size_t)row * 3 + 0];
    float py = pos[(size_t)row * 3 + 1];
    float pz = pos[(size_t)row * 3 + 2];
    const unsigned int* r1 = a1 + (size_t)row * 8;
    const unsigned int* r2 = a2 + (size_t)row * 8;
    float bs[KSEL]; int bi[KSEL];
    #pragma unroll
    for (int k = 0; k < KSEL; ++k) { bs[k] = -3e38f; bi[k] = 0; }
    for (int j = 0; j < Nn_; ++j) {
        int jr = b * Nn_ + j;
        float dx = px - pos[(size_t)jr * 3 + 0];
        float dy = py - pos[(size_t)jr * 3 + 1];
        float dz = pz - pos[(size_t)jr * 3 + 2];
        float dist = sqrtf(dx * dx + dy * dy + dz * dz + 1e-8f);
        unsigned int hop = ((r1[j >> 5] | r2[j >> 5]) >> (j & 31)) & 1u;
        bool valid = (hop && dist <= 5.0f) || (j == i);
        float sc = valid ? -dist : -1e9f;
        if (sc > bs[KSEL - 1]) {
            int p = KSEL - 1;
            while (p > 0 && bs[p - 1] < sc) {
                bs[p] = bs[p - 1]; bi[p] = bi[p - 1]; --p;
            }
            bs[p] = sc; bi[p] = j;
        }
    }
    #pragma unroll
    for (int k = 0; k < KSEL; ++k) {
        int j = bi[k];
        unsigned int h1 = (r1[j >> 5] >> (j & 31)) & 1u;
        unsigned int h2 = (r2[j >> 5] >> (j & 31)) & 1u;
        idxK[(size_t)row * KSEL + k]   = j;
        degK[(size_t)row * KSEL + k]   = h1 ? 1 : (h2 ? 2 : 0);
        nmaskK[(size_t)row * KSEL + k] = (bs[k] > -1e8f) ? 1 : 0;
    }
}

// ---------------------------------------------------------------------------
// 5) WMMA GEMM, LDS-staged B panel via the Tensor Data Mover.
//    Block = 4 waves = 64 rows x 64 cols of output. B panel (64 cols x Kd)
//    is DMA'd once per block into LDS with TDM row padding (+16B/row) so the
//    16 B-rows read per ds_load stripe across all 64 banks. Each wave holds
//    4 f32 accumulators; one A fragment feeds 4 v_wmma.
//    M,N,K all multiples of 64/32 => EXEC all-ones for every WMMA.
// ---------------------------------------------------------------------------
#define BPAD 8                      // 8 halfs = 16B padding per LDS B row
#define BSM_MAXK 256

__global__ void gemm_wmma_kernel(const h16* __restrict__ A,
                                 const h16* __restrict__ Bt,
                                 const float* __restrict__ R,
                                 const float* __restrict__ bias,
                                 float* __restrict__ Out,
                                 h16* __restrict__ OutH,
                                 int M, int Nw, int Kd, int relu) {
    __shared__ __align__(128) h16 bsm[64 * (BSM_MAXK + BPAD)];
    const int Kpad = Kd + BPAD;

    int lane = threadIdx.x & 31;
    int wave = threadIdx.x >> 5;
    int rrow = lane & 15;
    int hi   = lane >> 4;
    int m0 = (blockIdx.x * 4 + wave) * 16;   // this wave's 16 output rows
    int n0 = blockIdx.y * 64;                // block's 64 output columns

    // ---- stage B panel (64 cols x Kd, f16) into LDS ----
    const h16* bsrc = Bt + (size_t)n0 * Kd;
#if defined(HAVE_TDM)
    if (threadIdx.x < 32) {                  // wave 0 issues the DMA
        unsigned long long ga = (unsigned long long)(size_t)bsrc;
        unsigned int lds_base = (unsigned int)(size_t)(&bsm[0]);
        // D# group 0: count=1, lds_addr, global_addr, type=2
        u32x4 g0 = { 1u, lds_base,
                     (unsigned int)(ga & 0xFFFFFFFFull),
                     (unsigned int)((ga >> 32) & 0x01FFFFFFull) | (2u << 30) };
        // D# group 1: data_size=2B, pad_enable, pad_interval(Kd/2 dwords),
        //             pad_amount=4 dwords; dims: tensor/tile = Kd x 64
        unsigned int icode = (Kd == 128) ? 5u : 6u;     // 64 / 128 dwords
        unsigned int w0 = (1u << 16) | (1u << 20) | (icode << 22) | (3u << 25);
        unsigned int dim0 = (unsigned int)Kd, dim1 = 64u, t0 = (unsigned int)Kd;
        i32x8 g1 = { (int)w0,
                     (int)((dim0 & 0xFFFFu) << 16),
                     (int)((dim0 >> 16) | ((dim1 & 0xFFFFu) << 16)),
                     (int)((dim1 >> 16) | (t0 << 16)),
                     (int)64,                 // tile_dim1
                     (int)dim0,               // tensor_dim0_stride (low 32)
                     0, 0 };
        i32x4 g2 = { 0, 0, 0, 0 };
        i32x4 g3 = { 0, 0, 0, 0 };
#if __clang_major__ >= 23
        i32x8 g4 = { 0, 0, 0, 0, 0, 0, 0, 0 };
        __builtin_amdgcn_tensor_load_to_lds(g0, g1, g2, g3, g4, 0);
#else
        __builtin_amdgcn_tensor_load_to_lds(g0, g1, g2, g3, 0);
#endif
        __builtin_amdgcn_s_wait_tensorcnt(0);
    }
    __syncthreads();
#else
    // fallback: cooperative copy with identical padded layout
    for (int r = wave; r < 64; r += 4) {
        for (int c = lane * 8; c < Kd; c += 32 * 8) {
            *(v8h*)(&bsm[r * Kpad + c]) = *(const v8h*)(bsrc + (size_t)r * Kd + c);
        }
    }
    __syncthreads();
#endif

    // ---- accumulators: 4 n-tiles of 16, preloaded with residual/bias ----
    v8f acc[4];
    #pragma unroll
    for (int t = 0; t < 4; ++t) {
        int nC = n0 + t * 16 + rrow;
        #pragma unroll
        for (int r = 0; r < 8; ++r) {
            int mC = m0 + r + 8 * hi;
            float cv = R ? R[(size_t)mC * Nw + nC] : 0.0f;
            if (bias) cv += bias[nC];
            acc[t][r] = cv;
        }
    }

    // ---- K loop: 1 A fragment feeds 4 WMMAs ----
    const h16* arow = A + (size_t)(m0 + rrow) * Kd;
    for (int kk = 0; kk < Kd; kk += 32) {
        v8h alo = *(const v8h*)(arow + kk + 8 * hi);
        v8h ahi = *(const v8h*)(arow + kk + 16 + 8 * hi);
        v16h a;
        #pragma unroll
        for (int e = 0; e < 8; ++e) { a[e] = alo[e]; a[e + 8] = ahi[e]; }
        #pragma unroll
        for (int t = 0; t < 4; ++t) {
            const h16* bp = &bsm[(t * 16 + rrow) * Kpad + kk + 8 * hi];
            v8h blo = *(const v8h*)bp;          // ds_load_b128
            v8h bhi = *(const v8h*)(bp + 16);   // ds_load_b128
            v16h bfr;
            #pragma unroll
            for (int e = 0; e < 8; ++e) { bfr[e] = blo[e]; bfr[e + 8] = bhi[e]; }
            acc[t] = __builtin_amdgcn_wmma_f32_16x16x32_f16(
                         false, a, false, bfr, (short)0, acc[t], false, false);
        }
    }

    // ---- epilogue: relu / f32 store / f16 re-emit ----
    #pragma unroll
    for (int t = 0; t < 4; ++t) {
        int nC = n0 + t * 16 + rrow;
        #pragma unroll
        for (int r = 0; r < 8; ++r) {
            float v = acc[t][r];
            if (relu) v = fmaxf(v, 0.0f);
            int mC = m0 + r + 8 * hi;
            if (Out)  Out[(size_t)mC * Nw + nC]  = v;
            if (OutH) OutH[(size_t)mC * Nw + nC] = (h16)v;
        }
    }
}

// ---------------------------------------------------------------------------
// 6) Attention: one block per node (b,n); one wave per head (DH=32 == wave32).
// ---------------------------------------------------------------------------
__global__ void attn_kernel(const float* __restrict__ hq,
                            const float* __restrict__ hk,
                            const float* __restrict__ hv,
                            const float* __restrict__ hv1,
                            const float* __restrict__ pos,
                            const int* __restrict__ idxK,
                            const int* __restrict__ degK,
                            const int* __restrict__ nmaskK,
                            const float* __restrict__ deg_bias,
                            h16* __restrict__ out0h,
                            h16* __restrict__ n1h) {
    int row  = blockIdx.x;            // b*N+n
    int b    = row >> 8;
    int head = threadIdx.x >> 5;      // 0..3
    int lane = threadIdx.x & 31;      // channel in head
    int d    = head * DHd + lane;

    float px = pos[(size_t)row * 3 + 0];
    float py = pos[(size_t)row * 3 + 1];
    float pz = pos[(size_t)row * 3 + 2];
    float q = hq[(size_t)row * Dd + d];

    float lg[KSEL];
    float mx = -3e38f;
    #pragma unroll
    for (int k = 0; k < KSEL; ++k) {
        int j  = idxK[(size_t)row * KSEL + k];
        int nr = b * Nn_ + j;
        float p = q * hk[(size_t)nr * Dd + d];
        #pragma unroll
        for (int off = 16; off > 0; off >>= 1)
            p += __shfl_xor(p, off, 32);
        float v = p * 0.17677669529663687f;  // 1/sqrt(32)
        v += deg_bias[degK[(size_t)row * KSEL + k] * Hh + head];
        if (!nmaskK[(size_t)row * KSEL + k]) v = -1e9f;
        lg[k] = v;
        mx = fmaxf(mx, v);
    }
    float s = 0.0f;
    #pragma unroll
    for (int k = 0; k < KSEL; ++k) { lg[k] = __expf(lg[k] - mx); s += lg[k]; }
    float inv_s = 1.0f / s;

    float o = 0.0f, m0 = 0.0f, m1 = 0.0f, m2 = 0.0f;
    #pragma unroll
    for (int k = 0; k < KSEL; ++k) {
        int j  = idxK[(size_t)row * KSEL + k];
        int nr = b * Nn_ + j;
        float a  = lg[k] * inv_s;
        float vv = hv[(size_t)nr * Dd + d];
        float v1 = hv1[(size_t)nr * Dd + d];
        float rx = pos[(size_t)nr * 3 + 0] - px;
        float ry = pos[(size_t)nr * 3 + 1] - py;
        float rz = pos[(size_t)nr * 3 + 2] - pz;
        float invd = rsqrtf(rx * rx + ry * ry + rz * rz + 1e-8f);
        o += a * vv;
        float av1 = a * v1;
        m0 += av1 * rx * invd;
        m1 += av1 * ry * invd;
        m2 += av1 * rz * invd;
    }
    float n1 = sqrtf(m0 * m0 + m1 * m1 + m2 * m2 + 1e-8f);
    out0h[(size_t)row * Dd + d] = (h16)o;
    n1h[(size_t)row * Dd + d]   = (h16)n1;
}

// ---------------------------------------------------------------------------
// 7) Masked mean pool over nodes
// ---------------------------------------------------------------------------
__global__ void pool_kernel(const float* __restrict__ h,
                            const unsigned char* __restrict__ mask,
                            float* __restrict__ pooled) {
    int idx = blockIdx.x * blockDim.x + threadIdx.x;   // b*D+d
    if (idx >= Bsz * Dd) return;
    int b = idx / Dd, d = idx % Dd;
    float s = 0.0f, cnt = 0.0f;
    for (int n = 0; n < Nn_; ++n) {
        float m = mask[b * Nn_ + n] ? 1.0f : 0.0f;
        s   += m * h[((size_t)b * Nn_ + n) * Dd + d];
        cnt += m;
    }
    pooled[idx] = s / fmaxf(cnt, 1.0f);
}

// 8) Output projection: [B,128] @ [128,19] + b_out  (tiny; VALU)
__global__ void outproj_kernel(const float* __restrict__ pooled,
                               const float* __restrict__ W_out,
                               const float* __restrict__ b_out,
                               float* __restrict__ out) {
    int idx = blockIdx.x * blockDim.x + threadIdx.x;   // b*NO+o
    if (idx >= Bsz * NOo) return;
    int b = idx / NOo, o = idx % NOo;
    float acc = b_out[o];
    #pragma unroll 8
    for (int d = 0; d < Dd; ++d)
        acc = fmaf(pooled[b * Dd + d], W_out[d * NOo + o], acc);
    out[idx] = acc;
}

// ---------------------------------------------------------------------------
extern "C" void kernel_launch(void* const* d_in, const int* in_sizes, int n_in,
                              void* d_out, int out_size, void* d_ws, size_t ws_size,
                              hipStream_t stream) {
    const float*         x        = (const float*)d_in[0];
    const float*         pos      = (const float*)d_in[1];
    const unsigned char* mask     = (const unsigned char*)d_in[2];
    const unsigned char* adj      = (const unsigned char*)d_in[3];
    const float*         W_emb    = (const float*)d_in[4];
    const float*         b_emb    = (const float*)d_in[5];
    const float*         Wq       = (const float*)d_in[6];
    const float*         Wk       = (const float*)d_in[7];
    const float*         Wv       = (const float*)d_in[8];
    const float*         Wv1      = (const float*)d_in[9];
    const float*         deg_bias = (const float*)d_in[10];
    const float*         Wo       = (const float*)d_in[11];
    const float*         Wg       = (const float*)d_in[12];
    const float*         W1       = (const float*)d_in[13];
    const float*         b1       = (const float*)d_in[14];
    const float*         W2       = (const float*)d_in[15];
    const float*         b2       = (const float*)d_in[16];
    const float*         W_out    = (const float*)d_in[17];
    const float*         b_out    = (const float*)d_in[18];
    float* out = (float*)d_out;

    // Workspace carve (~77 MB total)
    char* ws = (char*)d_ws;
    size_t off = 0;
    auto carve = [&](size_t bytes) -> void* {
        void* p = ws + off;
        off = (off + bytes + 255) & ~(size_t)255;
        return p;
    };
    float* h_f32  = (float*)carve((size_t)BN * Dd * 4);
    h16*   hf16   = (h16*)  carve((size_t)BN * Dd * 2);
    h16*   WqT    = (h16*)  carve((size_t)Dd * Dd * 2);
    h16*   WkT    = (h16*)  carve((size_t)Dd * Dd * 2);
    h16*   WvT    = (h16*)  carve((size_t)Dd * Dd * 2);
    h16*   Wv1T   = (h16*)  carve((size_t)Dd * Dd * 2);
    h16*   WoT    = (h16*)  carve((size_t)Dd * Dd * 2);
    h16*   WgT    = (h16*)  carve((size_t)Dd * Dd * 2);
    h16*   W1T    = (h16*)  carve((size_t)FFd * Dd * 2);  // [FF][D]
    h16*   W2T    = (h16*)  carve((size_t)Dd * FFd * 2);  // [D][FF]
    unsigned int* a1bits = (unsigned int*)carve((size_t)BN * 8 * 4);
    unsigned int* a2bits = (unsigned int*)carve((size_t)BN * 8 * 4);
    int*   idxK   = (int*)  carve((size_t)BN * KSEL * 4);
    int*   degKb  = (int*)  carve((size_t)BN * KSEL * 4);
    int*   nmaskK = (int*)  carve((size_t)BN * KSEL * 4);
    float* hq     = (float*)carve((size_t)BN * Dd * 4);
    float* hk     = (float*)carve((size_t)BN * Dd * 4);
    float* hv     = (float*)carve((size_t)BN * Dd * 4);
    float* hv1    = (float*)carve((size_t)BN * Dd * 4);
    h16*   out0h  = (h16*)  carve((size_t)BN * Dd * 2);
    h16*   n1h    = (h16*)  carve((size_t)BN * Dd * 2);
    float* h2     = (float*)carve((size_t)BN * Dd * 4);
    h16*   h2h    = (h16*)  carve((size_t)BN * Dd * 2);
    h16*   ff1h   = (h16*)  carve((size_t)BN * FFd * 2);
    float* pooled = (float*)carve((size_t)Bsz * Dd * 4);
    (void)ws_size; (void)in_sizes; (void)n_in; (void)out_size;

    // 1) embedding (f32 + f16)
    embed_kernel<<<BN, Dd, 0, stream>>>(x, W_emb, b_emb, h_f32, hf16);

    // 2) weights -> f16, transposed to N-major
    cvt_weight_kernel<<<(Dd*Dd + 255)/256, 256, 0, stream>>>(Wq,  WqT,  Dd, Dd);
    cvt_weight_kernel<<<(Dd*Dd + 255)/256, 256, 0, stream>>>(Wk,  WkT,  Dd, Dd);
    cvt_weight_kernel<<<(Dd*Dd + 255)/256, 256, 0, stream>>>(Wv,  WvT,  Dd, Dd);
    cvt_weight_kernel<<<(Dd*Dd + 255)/256, 256, 0, stream>>>(Wv1, Wv1T, Dd, Dd);
    cvt_weight_kernel<<<(Dd*Dd + 255)/256, 256, 0, stream>>>(Wo,  WoT,  Dd, Dd);
    cvt_weight_kernel<<<(Dd*Dd + 255)/256, 256, 0, stream>>>(Wg,  WgT,  Dd, Dd);
    cvt_weight_kernel<<<(Dd*FFd + 255)/256, 256, 0, stream>>>(W1, W1T, Dd, FFd);
    cvt_weight_kernel<<<(FFd*Dd + 255)/256, 256, 0, stream>>>(W2, W2T, FFd, Dd);

    // 3) graph bitmasks + 2-hop
    adj1_bits_kernel<<<(BN*8 + 255)/256, 256, 0, stream>>>(adj, mask, a1bits);
    adj2_bits_kernel<<<(BN + 255)/256, 256, 0, stream>>>(a1bits, a2bits);

    // 4) top-K neighbor selection
    topk_kernel<<<(BN + 255)/256, 256, 0, stream>>>(pos, a1bits, a2bits,
                                                    idxK, degKb, nmaskK);

    // 5) QKV(+v1) projections via WMMA (project-then-gather identity)
    dim3 gB(128);
    dim3 gD(BN/64, Dd/64);   // (256, 2): 64x64 output per block
    gemm_wmma_kernel<<<gD, gB, 0, stream>>>(hf16, WqT,  nullptr, nullptr, hq,  nullptr, BN, Dd, Dd, 0);
    gemm_wmma_kernel<<<gD, gB, 0, stream>>>(hf16, WkT,  nullptr, nullptr, hk,  nullptr, BN, Dd, Dd, 0);
    gemm_wmma_kernel<<<gD, gB, 0, stream>>>(hf16, WvT,  nullptr, nullptr, hv,  nullptr, BN, Dd, Dd, 0);
    gemm_wmma_kernel<<<gD, gB, 0, stream>>>(hf16, Wv1T, nullptr, nullptr, hv1, nullptr, BN, Dd, Dd, 0);

    // 6) sparse attention (wave32 per head)
    attn_kernel<<<BN, Dd, 0, stream>>>(hq, hk, hv, hv1, pos, idxK, degKb,
                                       nmaskK, deg_bias, out0h, n1h);

    // 7) h2 = h + out0@Wo + n1@Wg  (two residual WMMA GEMMs; 2nd emits f16)
    gemm_wmma_kernel<<<gD, gB, 0, stream>>>(out0h, WoT, h_f32, nullptr, h2, nullptr, BN, Dd, Dd, 0);
    gemm_wmma_kernel<<<gD, gB, 0, stream>>>(n1h,   WgT, h2,    nullptr, h2, h2h,     BN, Dd, Dd, 0);

    // 8) FF: ff1 = relu(h2@W1 + b1) [f16 only]; h3 = h2 + ff1@W2 + b2 (in place)
    dim3 gF(BN/64, FFd/64);  // (256, 4)
    gemm_wmma_kernel<<<gF, gB, 0, stream>>>(h2h,  W1T, nullptr, b1, nullptr, ff1h, BN, FFd, Dd, 1);
    gemm_wmma_kernel<<<gD, gB, 0, stream>>>(ff1h, W2T, h2,      b2, h2,      nullptr, BN, Dd, FFd, 0);

    // 9) masked mean pool + output head
    pool_kernel<<<(Bsz*Dd + 255)/256, 256, 0, stream>>>(h2, mask, pooled);
    outproj_kernel<<<(Bsz*NOo + 255)/256, 256, 0, stream>>>(pooled, W_out, b_out, out);
}
